// PairwiseAttentionLinear_76871324664264
// MI455X (gfx1250) — matrix operand order repaired and checked by reading, not compiled, not run on hardware
//
#include <hip/hip_runtime.h>

// ---------------------------------------------------------------------------
// Types / helpers
// ---------------------------------------------------------------------------
typedef unsigned short u16;   // bf16 storage
typedef __attribute__((ext_vector_type(16))) __bf16 v16bf;
typedef __attribute__((ext_vector_type(8)))  float  v8f;
typedef int v4i __attribute__((vector_size(16)));   // matches builtin param type

union BFrag {
  v16bf v;
  uint4 q[2];
  u16   s[16];
};

__device__ __forceinline__ u16 f32_to_bf16(float f) {
  union { float f; unsigned int u; } c; c.f = f;
  unsigned int r = c.u + 0x7FFFu + ((c.u >> 16) & 1u);  // RNE
  return (u16)(r >> 16);
}
__device__ __forceinline__ float bf16_to_f32(u16 h) {
  union { unsigned int u; float f; } c; c.u = ((unsigned int)h) << 16;
  return c.f;
}

// ---- CDNA5 async global->LDS path (guarded; falls back to sync staging) ----
#if defined(__gfx1250__) && __has_builtin(__builtin_amdgcn_global_load_async_to_lds_b128)
#define USE_ASYNC 1
typedef __attribute__((address_space(1))) v4i* glb_v4i_ptr;
typedef __attribute__((address_space(3))) v4i* lds_v4i_ptr;
#else
#define USE_ASYNC 0
#endif

#if __has_builtin(__builtin_amdgcn_s_wait_asynccnt)
#define WAIT_ASYNC(n) __builtin_amdgcn_s_wait_asynccnt(n)
#else
#define WAIT_ASYNC(n) asm volatile("s_wait_asynccnt %0" ::"i"(n) : "memory")
#endif

// ---------------------------------------------------------------------------
// bf16 WMMA GEMM:  C[M,N] = A[M,K] * B[N,K]^T  (+bias, relu, residual)
//   A, B: bf16 row-major (lda/ldb = K-stride)
//   flags: 1=bias  2=relu  4=residual-add  8=store bf16 (else f32)
// Block: 256 threads (8 waves).  Tile: 128x128, BK=64, double-buffered LDS.
// ---------------------------------------------------------------------------
#define BM   128
#define BN   128
#define BKK  64
#define LDSS 72   // ushorts per LDS row (144B, multiple of 16B)

__global__ __launch_bounds__(256)
void gemm_bf16_wmma(const u16* __restrict__ A, const u16* __restrict__ B,
                    void* __restrict__ C, int M, int N, int K,
                    int lda, int ldb, int ldc,
                    const float* __restrict__ bias,
                    const float* __restrict__ resid, int ldr,
                    int flags)
{
#if USE_ASYNC
  __shared__ u16 As[2 * BM * LDSS];
  __shared__ u16 Bs[2 * BN * LDSS];
#else
  __shared__ u16 As[BM * LDSS];
  __shared__ u16 Bs[BN * LDSS];
#endif

  const int tid  = threadIdx.x;
  const int m0   = blockIdx.y * BM;
  const int n0   = blockIdx.x * BN;
  const int w    = tid >> 5;
  const int lane = tid & 31;
  const int wm   = (w & 3) * 32;   // wave row offset within tile
  const int wn   = (w >> 2) * 64;  // wave col offset within tile
  const int lhi  = lane >> 4;      // 0/1 half-wave
  const int llo  = lane & 15;

  v8f acc[2][4];
#pragma unroll
  for (int i = 0; i < 2; ++i)
#pragma unroll
    for (int j = 0; j < 4; ++j) acc[i][j] = (v8f){};

  // one BK=64 stage (two 32-wide K-steps) of WMMA work from a staged LDS buffer
  auto compute = [&](const u16* Ac, const u16* Bc) {
#pragma unroll
    for (int ks = 0; ks < 2; ++ks) {
      const int kofs = ks * 32;
      BFrag af[2], bf[4];
#pragma unroll
      for (int ti = 0; ti < 2; ++ti) {
        const u16* ap = Ac + (wm + ti * 16 + llo) * LDSS + kofs;
        af[ti].q[0] = *(const uint4*)(ap + lhi * 8);        // K 0..7 / 8..15
        af[ti].q[1] = *(const uint4*)(ap + 16 + lhi * 8);   // K 16..23 / 24..31
      }
#pragma unroll
      for (int tj = 0; tj < 4; ++tj) {
        const u16* bp = Bc + (wn + tj * 16 + llo) * LDSS + kofs;
        bf[tj].q[0] = *(const uint4*)(bp + lhi * 8);
        bf[tj].q[1] = *(const uint4*)(bp + 16 + lhi * 8);
      }
#pragma unroll
      for (int ti = 0; ti < 2; ++ti)
#pragma unroll
        for (int tj = 0; tj < 4; ++tj)
          acc[ti][tj] = __builtin_amdgcn_wmma_f32_16x16x32_bf16(
              false, af[ti].v, false, bf[tj].v,
              (short)0, acc[ti][tj], false, false);
    }
  };

#if USE_ASYNC
  // issue one stage (128x64 of A and B) of async global->LDS copies:
  // 8 b128 ops per thread -> ASYNCcnt +8 per wave.
  auto issue = [&](int kb, u16* Ad, u16* Bd) {
#pragma unroll
    for (int r = 0; r < 4; ++r) {
      int li  = tid + r * 256;        // 0..1023
      int row = li >> 3;              // 0..127
      int cb  = (li & 7) * 8;         // 0..56
      __builtin_amdgcn_global_load_async_to_lds_b128(
          (glb_v4i_ptr)(A + (size_t)(m0 + row) * lda + kb + cb),
          (lds_v4i_ptr)(Ad + row * LDSS + cb),
          0, 0);
      __builtin_amdgcn_global_load_async_to_lds_b128(
          (glb_v4i_ptr)(B + (size_t)(n0 + row) * ldb + kb + cb),
          (lds_v4i_ptr)(Bd + row * LDSS + cb),
          0, 0);
    }
  };

  issue(0, As, Bs);  // prologue: stage 0
  for (int kb = 0; kb < K; kb += BKK) {
    const int cur = (kb >> 6) & 1;
    const u16* Ac = As + cur * (BM * LDSS);
    const u16* Bc = Bs + cur * (BN * LDSS);
    if (kb + BKK < K) {
      issue(kb + BKK, As + (cur ^ 1) * (BM * LDSS),
                      Bs + (cur ^ 1) * (BN * LDSS));
      WAIT_ASYNC(8);   // 8 older ops (current stage) complete; next in flight
    } else {
      WAIT_ASYNC(0);
    }
    __syncthreads();   // current stage visible to all waves
    compute(Ac, Bc);
    __syncthreads();   // all waves done reading before buffer reuse
  }
#else
  for (int kb = 0; kb < K; kb += BKK) {
#pragma unroll
    for (int r = 0; r < 4; ++r) {
      int li  = tid + r * 256;
      int row = li >> 3;
      int cb  = (li & 7) * 8;
      *(uint4*)(As + row * LDSS + cb) =
          *(const uint4*)(A + (size_t)(m0 + row) * lda + kb + cb);
      *(uint4*)(Bs + row * LDSS + cb) =
          *(const uint4*)(B + (size_t)(n0 + row) * ldb + kb + cb);
    }
    __syncthreads();
    compute(As, Bs);
    __syncthreads();
  }
#endif

  // epilogue
#pragma unroll
  for (int ti = 0; ti < 2; ++ti) {
#pragma unroll
    for (int tj = 0; tj < 4; ++tj) {
      const int col  = n0 + wn + tj * 16 + llo;
      const float bv = (flags & 1) ? bias[col] : 0.0f;
#pragma unroll
      for (int e = 0; e < 8; ++e) {
        const int row = m0 + wm + ti * 16 + lhi * 8 + e;
        float val = acc[ti][tj][e] + bv;
        if (flags & 2) val = fmaxf(val, 0.0f);
        if (flags & 4) val += resid[(size_t)row * ldr + col];
        if (flags & 8) ((u16*)C)[(size_t)row * ldc + col] = f32_to_bf16(val);
        else           ((float*)C)[(size_t)row * ldc + col] = val;
      }
    }
  }
}

// ---------------------------------------------------------------------------
// Row LayerNorm (f32 in, bf16 out).  One 256-thread block per row.
// ---------------------------------------------------------------------------
__global__ __launch_bounds__(256)
void ln_bf16_kernel(const float* __restrict__ x, int ldx, int ncols,
                    const float* __restrict__ g, const float* __restrict__ b,
                    u16* __restrict__ out, int ldo)
{
  __shared__ float red0[8], red1[8];
  const int row = blockIdx.x;
  const float* xr = x + (size_t)row * ldx;

  float s = 0.0f, s2 = 0.0f;
  for (int c = threadIdx.x; c < ncols; c += 256) {
    float v = xr[c]; s += v; s2 += v * v;
  }
#pragma unroll
  for (int off = 16; off > 0; off >>= 1) {
    s  += __shfl_xor(s,  off, 32);
    s2 += __shfl_xor(s2, off, 32);
  }
  const int wv = threadIdx.x >> 5, lane = threadIdx.x & 31;
  if (lane == 0) { red0[wv] = s; red1[wv] = s2; }
  __syncthreads();
  if (threadIdx.x == 0) {
    float a = 0.0f, a2 = 0.0f;
#pragma unroll
    for (int i = 0; i < 8; ++i) { a += red0[i]; a2 += red1[i]; }
    red0[0] = a; red1[0] = a2;
  }
  __syncthreads();
  const float inv  = 1.0f / (float)ncols;
  const float mean = red0[0] * inv;
  const float var  = red1[0] * inv - mean * mean;
  const float rstd = rsqrtf(var + 1e-6f);

  u16* orow = out + (size_t)row * ldo;
  for (int c = threadIdx.x; c < ncols; c += 256)
    orow[c] = f32_to_bf16((xr[c] - mean) * rstd * g[c] + b[c]);
}

// ---------------------------------------------------------------------------
// f32 -> bf16 cast (vectorized)
// ---------------------------------------------------------------------------
__global__ __launch_bounds__(256)
void cast_f32_bf16(const float* __restrict__ in, u16* __restrict__ out, int n)
{
  int i = (blockIdx.x * 256 + threadIdx.x) * 4;
  if (i < n) {
    float4 f = *(const float4*)(in + i);
    out[i + 0] = f32_to_bf16(f.x);
    out[i + 1] = f32_to_bf16(f.y);
    out[i + 2] = f32_to_bf16(f.z);
    out[i + 3] = f32_to_bf16(f.w);
  }
}

// ---------------------------------------------------------------------------
// Seq-len-2 per-head attention + residual (in-place on x).
// One wave32 per (row, head).  DK=128 -> 4 elements per lane.
// ---------------------------------------------------------------------------
__global__ __launch_bounds__(256)
void attn2_kernel(const float* __restrict__ q, const float* __restrict__ k,
                  const float* __restrict__ v, float* __restrict__ x, int nrows)
{
  const int gw   = (int)((blockIdx.x * 256 + threadIdx.x) >> 5);
  const int lane = threadIdx.x & 31;
  const int row  = gw >> 3;
  const int head = gw & 7;
  if (row >= nrows) return;

  const size_t base = (size_t)row * 2048 + (size_t)head * 256;
  float s00 = 0.f, s01 = 0.f, s10 = 0.f, s11 = 0.f;
#pragma unroll
  for (int i = 0; i < 4; ++i) {
    const int d = lane + i * 32;
    float q0 = q[base + d], q1 = q[base + 128 + d];
    float k0 = k[base + d], k1 = k[base + 128 + d];
    s00 += q0 * k0; s01 += q0 * k1; s10 += q1 * k0; s11 += q1 * k1;
  }
#pragma unroll
  for (int off = 16; off > 0; off >>= 1) {
    s00 += __shfl_xor(s00, off, 32);
    s01 += __shfl_xor(s01, off, 32);
    s10 += __shfl_xor(s10, off, 32);
    s11 += __shfl_xor(s11, off, 32);
  }
  const float SCALE = 0.088388347648318447f;  // 1/sqrt(128)
  s00 *= SCALE; s01 *= SCALE; s10 *= SCALE; s11 *= SCALE;
  const float m0 = fmaxf(s00, s01), m1 = fmaxf(s10, s11);
  const float e00 = __expf(s00 - m0), e01 = __expf(s01 - m0);
  const float e10 = __expf(s10 - m1), e11 = __expf(s11 - m1);
  const float r0 = 1.0f / (e00 + e01), r1 = 1.0f / (e10 + e11);
  const float a00 = e00 * r0, a01 = e01 * r0;
  const float a10 = e10 * r1, a11 = e11 * r1;
#pragma unroll
  for (int i = 0; i < 4; ++i) {
    const int d = lane + i * 32;
    const float v0 = v[base + d], v1 = v[base + 128 + d];
    x[base + d]       += a00 * v0 + a01 * v1;
    x[base + 128 + d] += a10 * v0 + a11 * v1;
  }
}

// ---------------------------------------------------------------------------
// Final head: out[r] = dot(h[r, 0:1024](bf16), w2) + b2.  One wave per row.
// ---------------------------------------------------------------------------
__global__ __launch_bounds__(256)
void final_dot_kernel(const u16* __restrict__ h, const float* __restrict__ w2,
                      const float* __restrict__ b2, float* __restrict__ out,
                      int nrows)
{
  const int gw   = (int)((blockIdx.x * 256 + threadIdx.x) >> 5);
  const int lane = threadIdx.x & 31;
  if (gw >= nrows) return;
  const u16* hr = h + (size_t)gw * 1024;
  float s = 0.0f;
#pragma unroll 8
  for (int i = 0; i < 32; ++i) {
    const int c = lane + i * 32;
    s += bf16_to_f32(hr[c]) * w2[c];
  }
#pragma unroll
  for (int off = 16; off > 0; off >>= 1) s += __shfl_xor(s, off, 32);
  if (lane == 0) out[gw] = s + b2[0];
}

// ---------------------------------------------------------------------------
// Host orchestration
// ---------------------------------------------------------------------------
static inline void launch_gemm(const u16* A, const u16* B, void* C,
                               int M, int N, int K, int lda, int ldb, int ldc,
                               const float* bias, const float* resid, int ldr,
                               int flags, hipStream_t s)
{
  dim3 grid(N / BN, M / BM), blk(256);
  gemm_bf16_wmma<<<grid, blk, 0, s>>>(A, B, C, M, N, K, lda, ldb, ldc,
                                      bias, resid, ldr, flags);
}

extern "C" void kernel_launch(void* const* d_in, const int* in_sizes, int n_in,
                              void* d_out, int out_size, void* d_ws, size_t ws_size,
                              hipStream_t stream)
{
  constexpr int Bsz = 8192, Dm = 2048, HD = 1024, NLc = 3;
  (void)in_sizes; (void)n_in; (void)out_size; (void)ws_size;

  const float* x     = (const float*)d_in[0];
  const float* alng  = (const float*)d_in[1];
  const float* alnb  = (const float*)d_in[2];
  const float* Wq    = (const float*)d_in[3];
  const float* Wk    = (const float*)d_in[4];
  const float* Wv    = (const float*)d_in[5];
  const float* lndg  = (const float*)d_in[6];
  const float* lndb  = (const float*)d_in[7];
  const float* lntg  = (const float*)d_in[8];
  const float* lntb  = (const float*)d_in[9];
  const float* Wd1   = (const float*)d_in[10];
  const float* bd1   = (const float*)d_in[11];
  const float* Wd2   = (const float*)d_in[12];
  const float* bd2   = (const float*)d_in[13];
  const float* Wt1   = (const float*)d_in[14];
  const float* bt1   = (const float*)d_in[15];
  const float* Wt2   = (const float*)d_in[16];
  const float* bt2   = (const float*)d_in[17];
  const float* Wo1   = (const float*)d_in[18];
  const float* bo1   = (const float*)d_in[19];
  const float* Wo2   = (const float*)d_in[20];
  const float* bo2   = (const float*)d_in[21];

  // workspace layout (all 256B aligned)
  constexpr size_t XCUR = 0;                                  // f32  8192x2048
  constexpr size_t XN   = XCUR + (size_t)Bsz * Dm * 4;        // bf16 8192x2048
  constexpr size_t QB   = XN   + (size_t)Bsz * Dm * 2;        // f32  8192x2048
  constexpr size_t KB   = QB   + (size_t)Bsz * Dm * 4;
  constexpr size_t VB   = KB   + (size_t)Bsz * Dm * 4;
  constexpr size_t WB   = VB   + (size_t)Bsz * Dm * 4;        // bf16 2048x2048
  constexpr size_t H1   = WB   + (size_t)Dm * Dm * 2;         // bf16 8192x2048
  constexpr size_t DN   = H1   + (size_t)Bsz * Dm * 2;        // bf16 8192x1024
  constexpr size_t TN   = DN   + (size_t)Bsz * HD * 2;        // bf16 8192x1024

  char* ws = (char*)d_ws;
  float* xcur = (float*)(ws + XCUR);
  u16*   xn   = (u16*)(ws + XN);
  float* qb   = (float*)(ws + QB);
  float* kbuf = (float*)(ws + KB);
  float* vbuf = (float*)(ws + VB);
  u16*   wb   = (u16*)(ws + WB);
  u16*   h1   = (u16*)(ws + H1);
  u16*   dn   = (u16*)(ws + DN);
  u16*   tn   = (u16*)(ws + TN);

  (void)hipMemcpyAsync(xcur, x, (size_t)Bsz * Dm * sizeof(float),
                       hipMemcpyDeviceToDevice, stream);

  const int castDD = (Dm * Dm / 4) / 256;   // 2048x2048 weight cast blocks
  const int castDH = (Dm * HD / 4) / 256;   // 2048x1024 weight cast blocks

  for (int n = 0; n < NLc; ++n) {
    // ---- attention layer ----
    ln_bf16_kernel<<<Bsz, 256, 0, stream>>>(xcur, Dm, Dm, alng + n * Dm,
                                            alnb + n * Dm, xn, Dm);
    cast_f32_bf16<<<castDD, 256, 0, stream>>>(Wq + (size_t)n * Dm * Dm, wb, Dm * Dm);
    launch_gemm(xn, wb, qb, Bsz, Dm, Dm, Dm, Dm, Dm, nullptr, nullptr, 0, 0, stream);
    cast_f32_bf16<<<castDD, 256, 0, stream>>>(Wk + (size_t)n * Dm * Dm, wb, Dm * Dm);
    launch_gemm(xn, wb, kbuf, Bsz, Dm, Dm, Dm, Dm, Dm, nullptr, nullptr, 0, 0, stream);
    cast_f32_bf16<<<castDD, 256, 0, stream>>>(Wv + (size_t)n * Dm * Dm, wb, Dm * Dm);
    launch_gemm(xn, wb, vbuf, Bsz, Dm, Dm, Dm, Dm, Dm, nullptr, nullptr, 0, 0, stream);
    attn2_kernel<<<(Bsz * 8 * 32) / 256, 256, 0, stream>>>(qb, kbuf, vbuf, xcur, Bsz);

    if (n == NLc - 1) break;

    // ---- feed-forward layer (split halves) ----
    ln_bf16_kernel<<<Bsz, 256, 0, stream>>>(xcur, Dm, HD, lndg + n * HD,
                                            lndb + n * HD, dn, HD);
    ln_bf16_kernel<<<Bsz, 256, 0, stream>>>(xcur + HD, Dm, HD, lntg + n * HD,
                                            lntb + n * HD, tn, HD);
    // d branch: relu(dn @ Wd1^T + bd1) @ Wd2^T + bd2 + resid -> xcur[:, :HD]
    cast_f32_bf16<<<castDH, 256, 0, stream>>>(Wd1 + (size_t)n * Dm * HD, wb, Dm * HD);
    launch_gemm(dn, wb, h1, Bsz, Dm, HD, HD, HD, Dm, bd1 + n * Dm,
                nullptr, 0, 1 | 2 | 8, stream);
    cast_f32_bf16<<<castDH, 256, 0, stream>>>(Wd2 + (size_t)n * HD * Dm, wb, HD * Dm);
    launch_gemm(h1, wb, xcur, Bsz, HD, Dm, Dm, Dm, Dm, bd2 + n * HD,
                xcur, Dm, 1 | 4, stream);
    // t branch -> xcur[:, HD:]
    cast_f32_bf16<<<castDH, 256, 0, stream>>>(Wt1 + (size_t)n * Dm * HD, wb, Dm * HD);
    launch_gemm(tn, wb, h1, Bsz, Dm, HD, HD, HD, Dm, bt1 + n * Dm,
                nullptr, 0, 1 | 2 | 8, stream);
    cast_f32_bf16<<<castDH, 256, 0, stream>>>(Wt2 + (size_t)n * HD * Dm, wb, HD * Dm);
    launch_gemm(h1, wb, xcur + HD, Bsz, HD, Dm, Dm, Dm, Dm, bt2 + n * HD,
                xcur + HD, Dm, 1 | 4, stream);
  }

  // ---- final head ----
  cast_f32_bf16<<<(Bsz * Dm / 4) / 256, 256, 0, stream>>>(xcur, xn, Bsz * Dm);
  cast_f32_bf16<<<castDH, 256, 0, stream>>>(Wo1, wb, HD * Dm);
  launch_gemm(xn, wb, dn, Bsz, HD, Dm, Dm, Dm, HD, bo1, nullptr, 0, 1 | 2 | 8, stream);
  final_dot_kernel<<<(Bsz * 32) / 256, 256, 0, stream>>>(dn, Wo2, bo2,
                                                         (float*)d_out, Bsz);
}
